// FusedModConv_68238440398863
// MI455X (gfx1250) — matrix operand order repaired and checked by reading, not compiled, not run on hardware
//
#include <hip/hip_runtime.h>

// ---------------- problem constants ----------------
#define B_      4
#define CIN     64
#define COUT    64
#define HH      512
#define WW      512
#define KK      3
#define WDIM    512
#define HW      (HH * WW)

// GEMM over K = 3*3*64 = 576, split into 18 chunks of 32 (kh,kw major, 2 cin halves)
#define NCHUNK  18
// weight fragment storage: per b: 4 mtiles * 18 chunks * 32 lanes * 16 bf16
#define WFRAG_PER_B (4 * NCHUNK * 32 * 16)

typedef __attribute__((ext_vector_type(16))) __bf16 v16bf;
typedef __attribute__((ext_vector_type(8)))  float  v8f;

static __device__ __forceinline__ unsigned short f2bf(float f) {
    unsigned int u = __float_as_uint(f);
    u += 0x7FFFu + ((u >> 16) & 1u);   // round-to-nearest-even
    return (unsigned short)(u >> 16);
}

// ---------------------------------------------------------------------------
// Kernel 1: style GEMM + modulate + demodulate + swizzle weights into
// WMMA A-fragment layout (16x32 bf16 A matrix, ISA 7.12.2) in d_ws.
// One block per batch sample.
// ---------------------------------------------------------------------------
__global__ void modweights_kernel(const float* __restrict__ w,
                                  const float* __restrict__ conv_w,
                                  const float* __restrict__ mod_w,
                                  const float* __restrict__ mod_b,
                                  unsigned short* __restrict__ wfrag) {
    const int b = blockIdx.x;
    const int t = threadIdx.x;

    __shared__ float style_s[CIN];
    __shared__ float dcoef_s[COUT];

    const float rc_dense = 1.0f / sqrtf((float)WDIM);
    const float rc_conv  = 1.0f / sqrtf((float)(KK * KK * CIN));

    // ---- style[b, cin] = w[b,:] @ (rc*mod_w) + mod_b + 1 ----
    if (t < CIN) {
        float s = 0.0f;
        const float* wb = w + b * WDIM;
        for (int i = 0; i < WDIM; ++i)
            s += wb[i] * mod_w[i * CIN + t];
        style_s[t] = s * rc_dense + mod_b[t] + 1.0f;
    }
    __syncthreads();

    // ---- demod coef per cout: rsqrt(sum_k (rc*cw*style)^2 + 1e-8), fold rc ----
    if (t < COUT) {
        float sum = 0.0f;
        for (int kk = 0; kk < KK * KK; ++kk)
            for (int ci = 0; ci < CIN; ++ci) {
                float v = conv_w[(kk * CIN + ci) * COUT + t] * style_s[ci];
                sum += v * v;
            }
        float d = rsqrtf(rc_conv * rc_conv * sum + 1e-8f);
        dcoef_s[t] = rc_conv * d;  // final weight = conv_w * style * dcoef
    }
    __syncthreads();

    // ---- write fragments: layout [mtile][chunk][lane][e], 36864 elems / b ----
    // A-matrix 16x32 bf16 layout: lane m = lane&15;
    //  element e: K = (lane<16 ? 0 : 16) + e  (contiguous, matches B fragment)
    unsigned short* wb = wfrag + b * WFRAG_PER_B;
    for (int f = t; f < WFRAG_PER_B; f += 256) {
        int e     = f & 15;
        int lane  = (f >> 4) & 31;
        int chunk = (f >> 9) % NCHUNK;
        int mtile = f / (NCHUNK * 512);

        int v = e >> 1, j = e & 1;
        int Kidx = ((lane < 16) ? 0 : 16) + 2 * v + j;

        int m    = lane & 15;
        int cout = mtile * 16 + m;
        int kh   = chunk / 6;
        int kw   = (chunk >> 1) % 3;
        int cin  = (chunk & 1) * 32 + Kidx;

        float wt = conv_w[((kh * 3 + kw) * CIN + cin) * COUT + cout]
                   * style_s[cin] * dcoef_s[cout];
        wb[f] = f2bf(wt);
    }
}

// ---------------------------------------------------------------------------
// Kernel 2: implicit-GEMM 3x3 grouped conv with v_wmma_f32_16x16x32_bf16.
// grid = (WW/32, HH/4, B). 256 threads = 8 waves.
// Wave wid: mtile = wid&3 (16 couts), whalf = wid>>2 (16 pixels), 4 rows.
// ---------------------------------------------------------------------------
#define TILE_W  32
#define TILE_H  4
#define LCOLS   (TILE_W + 2)   // 34
#define LROWS   (TILE_H + 2)   // 6

__global__ void __launch_bounds__(256)
modconv_kernel(const float* __restrict__ x,
               const unsigned short* __restrict__ wfrag,
               float* __restrict__ out) {
    const int b  = blockIdx.z;
    const int h0 = blockIdx.y * TILE_H;
    const int w0 = blockIdx.x * TILE_W;
    const int t  = threadIdx.x;

    // x tile in LDS: [row 0..5][col 0..33][cin 0..63] bf16  (cin contiguous)
    __shared__ unsigned short xs[LROWS * LCOLS * CIN];   // 26112 B

    // ---- stage x tile (halo + zero pad), convert f32 -> bf16 ----
    // 6*34*64 = 13056 elements; 13056/256 = 51 exact -> no divergence
    const float* xb = x + (size_t)b * CIN * HW;
    for (int i = t; i < LROWS * LCOLS * CIN; i += 256) {
        int cin = i / (LROWS * LCOLS);
        int pos = i % (LROWS * LCOLS);
        int row = pos / LCOLS;
        int col = pos % LCOLS;
        int gh = h0 - 1 + row;
        int gw = w0 - 1 + col;
        float v = 0.0f;
        if (gh >= 0 && gh < HH && gw >= 0 && gw < WW)
            v = xb[(size_t)cin * HW + gh * WW + gw];
        xs[(row * LCOLS + col) * CIN + cin] = f2bf(v);
    }
    __syncthreads();

    const int wid   = t >> 5;
    const int lane  = t & 31;
    const int mtile = wid & 3;
    const int whalf = wid >> 2;
    const int n     = lane & 15;
    const int kb    = (lane < 16) ? 0 : 16;   // B-frag K-half per lane group
    const int col_t = whalf * 16 + n;         // pixel column within tile

    v8f acc[TILE_H];
    #pragma unroll
    for (int r = 0; r < TILE_H; ++r) acc[r] = (v8f){};

    const unsigned short* wf =
        wfrag + ((size_t)b * 4 + mtile) * NCHUNK * 512;

    for (int c = 0; c < NCHUNK; ++c) {
        const int kh   = c / 6;
        const int kw   = (c >> 1) % 3;
        const int cinb = (c & 1) * 32;

        // A fragment: 32 B contiguous per lane (pre-swizzled in d_ws)
        v16bf a = *(const v16bf*)(wf + c * 512 + lane * 16);

        #pragma unroll
        for (int r = 0; r < TILE_H; ++r) {
            // B fragment: column = pixel (col_t+kw, row r+kh),
            // 16 consecutive cin starting at cinb+kb  -> 32 B from LDS
            const unsigned short* bp =
                &xs[((r + kh) * LCOLS + (col_t + kw)) * CIN + cinb + kb];
            v16bf bf = *(const v16bf*)bp;
            acc[r] = __builtin_amdgcn_wmma_f32_16x16x32_bf16(
                         false, a, false, bf, (short)0, acc[r], false, false);
        }
    }

    // ---- store: C/D layout: VGPR v -> M = v + (lane<16 ? 0 : 8), N = lane&15
    const int mbase = mtile * 16 + ((lane < 16) ? 0 : 8);
    #pragma unroll
    for (int r = 0; r < TILE_H; ++r) {
        #pragma unroll
        for (int v = 0; v < 8; ++v) {
            int cout = mbase + v;
            out[(((size_t)b * COUT + cout) * HH + (h0 + r)) * WW + (w0 + col_t)]
                = acc[r][v];
        }
    }
}

// ---------------------------------------------------------------------------
extern "C" void kernel_launch(void* const* d_in, const int* in_sizes, int n_in,
                              void* d_out, int out_size, void* d_ws, size_t ws_size,
                              hipStream_t stream) {
    const float* x      = (const float*)d_in[0];
    const float* w      = (const float*)d_in[1];
    const float* conv_w = (const float*)d_in[2];
    const float* mod_w  = (const float*)d_in[3];
    const float* mod_b  = (const float*)d_in[4];
    float* out          = (float*)d_out;
    unsigned short* wsw = (unsigned short*)d_ws;

    modweights_kernel<<<dim3(B_), 256, 0, stream>>>(w, conv_w, mod_w, mod_b, wsw);

    dim3 grid(WW / TILE_W, HH / TILE_H, B_);
    modconv_kernel<<<grid, 256, 0, stream>>>(x, wsw, out);
}
